// RNN_Flat_58806692217504
// MI455X (gfx1250) — matrix operand order repaired and checked by reading, not compiled, not run on hardware
//
#include <hip/hip_runtime.h>
#include <hip/hip_bf16.h>

// ---------------------------------------------------------------------------
// RNN_Flat on MI455X (gfx1250): persistent WMMA kernel.
//   Per cell:  h[l] = tanh( (prev + h_old[l]) @ Whh.T + (b_ih + b_bh) )
//   1536 sequential cells, grid of 16 WGs (one per WGP), 64 output columns
//   each, Whh slice resident in LDS as fp16, v_wmma_f32_16x16x32_f16 compute.
// ---------------------------------------------------------------------------

#define BATCH 64
#define HID   1024
#define SEQT  256
#define LAY   6
#define NWG   16

typedef _Float16 h16;
typedef __attribute__((ext_vector_type(16))) _Float16 v16h;
typedef __attribute__((ext_vector_type(8)))  _Float16 v8h;
typedef __attribute__((ext_vector_type(4)))  _Float16 v4h;
typedef __attribute__((ext_vector_type(8)))  float    v8f;

// ---------------------------------------------------------------- prep ------
__global__ void rnn_prep(const float* __restrict__ h0,
                         const float* __restrict__ Whh,
                         const float* __restrict__ bih,
                         const float* __restrict__ bbh,
                         float* __restrict__ hA,          // d_out + B*T  (h state buf A)
                         h16*   __restrict__ Wh16,
                         float* __restrict__ bias_sum,
                         unsigned* __restrict__ bar)
{
    const int i = blockIdx.x * blockDim.x + threadIdx.x;
    const int n = gridDim.x * blockDim.x;
    if (i == 0) bar[0] = 0u;
    for (int k = i; k < HID * HID; k += n) Wh16[k] = (h16)Whh[k];
    for (int k = i; k < LAY * BATCH * HID; k += n) hA[k] = h0[k];
    for (int k = i; k < HID; k += n) bias_sum[k] = bih[k] + bbh[k];
}

// ----------------------------------------------------------- persistent -----
__launch_bounds__(256, 1)
__global__ void rnn_persistent(float* __restrict__ hA,     // buffer A (== d_out h_final region)
                               float* __restrict__ hB,     // buffer B (scratch)
                               const h16* __restrict__ Wh, // fp16 Whh, row-major [H][H]
                               const float* __restrict__ bias,
                               const float* __restrict__ wout,
                               float* __restrict__ partial, // [T][B][NWG]
                               unsigned* __restrict__ bar)
{
    // LDS: B-slice 128KB + A(S) 128KB + 16.25KB tile + w  => ~273KB (< 320KB/WGP)
    __shared__ __align__(16) h16 Blds[64 * HID];   // Whh rows [g*64, g*64+64)
    __shared__ __align__(16) h16 Alds[64 * HID];   // S = prev + h_old[l], fp16
    __shared__ float tileout[64][65];              // padded: bank-conflict free
    __shared__ float wlds[64];

    const int g    = blockIdx.x;
    const int tid  = threadIdx.x;
    const int lane = tid & 31;
    const int wave = tid >> 5;
    const int l15  = lane & 15;
    const int hi   = (lane >> 4) & 1;              // half-wave select

    // Load resident Whh slice: rows g*64..g*64+63 (contiguous block).
    {
        const uint4* src = (const uint4*)(Wh + (size_t)g * 64 * HID);
        uint4*       dst = (uint4*)Blds;
        for (int i = tid; i < 64 * HID / 8; i += 256) dst[i] = src[i];
    }
    if (tid < 64) wlds[tid] = wout[g * 64 + tid];

    // Each of 8 waves: N-tile = wave%4, two M-tiles = {2*(wave/4), 2*(wave/4)+1}
    const int Nt   = wave & 3;
    const int Mt0  = (wave >> 2) << 1;
    const int colg = g * 64 + Nt * 16 + l15;       // this lane's output column
    const float biasc = bias[colg];
    __syncthreads();

    unsigned gen = 0;
    for (int t = 0; t < SEQT; ++t) {
        float* hOld = (t & 1) ? hB : hA;           // read side  (t=0 reads hA=h0)
        float* hNew = (t & 1) ? hA : hB;           // write side (t=255 writes hA)
        for (int l = 0; l < LAY; ++l) {
            const float* prev = (l == 0) ? (hOld + 5 * BATCH * HID)
                                         : (hNew + (l - 1) * BATCH * HID);
            const float* hl   = hOld + l * BATCH * HID;

            // Stage S = prev + h_old[l] into LDS as fp16 (row-major [64][1024]).
            {
                const float4* pa = (const float4*)prev;
                const float4* pb = (const float4*)hl;
                v4h* pd = (v4h*)Alds;
                for (int i = tid; i < BATCH * HID / 4; i += 256) {
                    float4 a = pa[i], b = pb[i];
                    v4h r = { (h16)(a.x + b.x), (h16)(a.y + b.y),
                              (h16)(a.z + b.z), (h16)(a.w + b.w) };
                    pd[i] = r;
                }
            }
            __syncthreads();

            // 64x64 output tile, K = 1024 in steps of 32 via WMMA f16->f32.
            v8f acc0 = {}, acc1 = {};
            const int arow0 = (Mt0 * 16 + l15) * HID;
            const int arow1 = arow0 + 16 * HID;
            const int brow  = (Nt * 16 + l15) * HID;
            for (int kk = 0; kk < HID; kk += 32) {
                // B frag: lane holds col N=l15, 16 consecutive K at K+16*hi (contiguous order, same as A).
                v16h bfrag = *(const v16h*)&Blds[brow + kk + hi * 16];
                // A frag: per-lane 16 consecutive K at K+16*hi (matches B).
                const int ka = kk + hi * 16;
                v8h a0lo = *(const v8h*)&Alds[arow0 + ka];
                v8h a0hi = *(const v8h*)&Alds[arow0 + ka + 8];
                v8h a1lo = *(const v8h*)&Alds[arow1 + ka];
                v8h a1hi = *(const v8h*)&Alds[arow1 + ka + 8];
                v16h a0, a1;
                #pragma unroll
                for (int q = 0; q < 8; ++q) {
                    a0[q] = a0lo[q]; a0[q + 8] = a0hi[q];
                    a1[q] = a1lo[q]; a1[q + 8] = a1hi[q];
                }
                acc0 = __builtin_amdgcn_wmma_f32_16x16x32_f16(
                           false, a0, false, bfrag, (short)0, acc0, false, false);
                acc1 = __builtin_amdgcn_wmma_f32_16x16x32_f16(
                           false, a1, false, bfrag, (short)0, acc1, false, false);
            }

            // Epilogue: bias + tanh, write new h[l]; stash top-layer for readout.
            float* outp = hNew + (size_t)l * BATCH * HID;
            #pragma unroll
            for (int r = 0; r < 8; ++r) {
                const int row0 = Mt0 * 16 + r + hi * 8;     // C layout: M=r+8*hi
                const int row1 = row0 + 16;
                const float v0 = tanhf(acc0[r] + biasc);
                const float v1 = tanhf(acc1[r] + biasc);
                outp[row0 * HID + colg] = v0;
                outp[row1 * HID + colg] = v1;
                if (l == LAY - 1) {
                    tileout[row0][Nt * 16 + l15] = v0;
                    tileout[row1][Nt * 16 + l15] = v1;
                }
            }
            if (l == LAY - 1) {                    // deterministic partial dot
                __syncthreads();
                if (tid < 64) {
                    float s = 0.f;
                    #pragma unroll 8
                    for (int c = 0; c < 64; ++c) s += tileout[tid][c] * wlds[c];
                    partial[((size_t)(t * BATCH + tid)) * NWG + g] = s;
                }
            }

            // Grid barrier: monotonic counter, release/acquire at device scope.
            ++gen;
            __threadfence();
            __syncthreads();
            if (tid == 0) {
                __hip_atomic_fetch_add(bar, 1u, __ATOMIC_RELEASE,
                                       __HIP_MEMORY_SCOPE_AGENT);
                while (__hip_atomic_load(bar, __ATOMIC_ACQUIRE,
                                         __HIP_MEMORY_SCOPE_AGENT) < gen * NWG) {
                    __builtin_amdgcn_s_sleep(2);
                }
            }
            __syncthreads();
            __threadfence();
        }
    }
}

// ------------------------------------------------------------- finalize -----
__global__ void rnn_finalize(const float* __restrict__ partial,
                             const float* __restrict__ b_out,
                             float* __restrict__ out)   // [B][T]
{
    const int j = blockIdx.x * blockDim.x + threadIdx.x;
    if (j >= BATCH * SEQT) return;
    const int b = j >> 8;          // j / T
    const int t = j & 255;         // j % T
    const float* p = partial + ((size_t)(t * BATCH + b)) * NWG;
    float s = b_out[0];
    #pragma unroll
    for (int g = 0; g < NWG; ++g) s += p[g];
    out[b * SEQT + t] = s;
}

// --------------------------------------------------------------- launch -----
extern "C" void kernel_launch(void* const* d_in, const int* in_sizes, int n_in,
                              void* d_out, int out_size, void* d_ws, size_t ws_size,
                              hipStream_t stream) {
    // inputs: 0=x(unused) 1=h0 2=weight_ih(unused) 3=bias_ih 4=weight_hh
    //         5=bias_bh 6=w_out 7=b_out     (all float32)
    const float* h0    = (const float*)d_in[1];
    const float* bih   = (const float*)d_in[3];
    const float* Whh   = (const float*)d_in[4];
    const float* bbh   = (const float*)d_in[5];
    const float* w_out = (const float*)d_in[6];
    const float* b_out = (const float*)d_in[7];

    float* out = (float*)d_out;                       // [B,T,1]   = 16384 floats
    float* hA  = (float*)d_out + BATCH * SEQT;        // [L,B,H]   = h_final region

    char* ws = (char*)d_ws;
    float*    hB       = (float*)(ws);                          // 1.5 MB
    h16*      Wh16     = (h16*)(ws + 1572864);                  // 2.0 MB
    float*    bias_sum = (float*)(ws + 1572864 + 2097152);      // 4 KB
    float*    partial  = (float*)(ws + 1572864 + 2097152 + 4096);       // 1 MB
    unsigned* bar      = (unsigned*)(ws + 1572864 + 2097152 + 4096 + 1048576);

    rnn_prep<<<256, 256, 0, stream>>>(h0, Whh, bih, bbh, hA, Wh16, bias_sum, bar);
    rnn_persistent<<<NWG, 256, 0, stream>>>(hA, hB, Wh16, bias_sum, w_out, partial, bar);
    rnn_finalize<<<(BATCH * SEQT + 255) / 256, 256, 0, stream>>>(partial, b_out, out);
}
